// GCN_73521250173546
// MI455X (gfx1250) — matrix operand order repaired and checked by reading, not compiled, not run on hardware
//
#include <hip/hip_runtime.h>
#include <hip/hip_bf16.h>
#include <math.h>

typedef _Float16 v16h __attribute__((ext_vector_type(16)));
typedef float    v8f  __attribute__((ext_vector_type(8)));
typedef float    f32x4 __attribute__((ext_vector_type(4)));

#define NN 100000
#define EE 3200000

// ---------------------------------------------------------------------------
// Pack W[K][Nout] (f32, row-major) into f16 WMMA B-fragments.
// Fragment (colTile ct, kTile kt) occupies 512 halfs: lane*16 + e, where
//   col = ct*16 + (lane&15),  k = kt*32 + (lane>=16 ? 8 : 0) + (e&7) + (e>=8 ? 16 : 0)
// (dense 32x16 f16 B layout matching A: half-wave holds K 0-7,16-23 / 8-15,24-31).
// ---------------------------------------------------------------------------
__global__ void pack_w_kernel(const float* __restrict__ W, _Float16* __restrict__ Wp,
                              int K, int Nout) {
  int idx = blockIdx.x * blockDim.x + threadIdx.x;
  int total = K * Nout;
  if (idx >= total) return;
  int e    = idx & 15;
  int lane = (idx >> 4) & 31;
  int frag = idx >> 9;
  int kSteps = K >> 5;
  int kt = frag % kSteps;
  int ct = frag / kSteps;
  int col = (ct << 4) + (lane & 15);
  int k   = (kt << 5) + ((lane >> 4) << 3) + (e & 7) + ((e >> 3) << 4);
  Wp[idx] = (_Float16)W[(size_t)k * Nout + col];
}

// Build an A-fragment (16x32 f16, ISA lane layout) from f32 row-major A.
// lane<16: halves 0..7 = K k0..k0+7, halves 8..15 = K k0+16..k0+23 (base 0)
// lane>=16: same with K base +8.
static __device__ __forceinline__ v16h load_a_frag(const float* __restrict__ Arow, int k0) {
  f32x4 a0 = *(const f32x4*)(Arow + k0);
  f32x4 a1 = *(const f32x4*)(Arow + k0 + 4);
  f32x4 a2 = *(const f32x4*)(Arow + k0 + 16);
  f32x4 a3 = *(const f32x4*)(Arow + k0 + 20);
  v16h af;
  af[0]=(_Float16)a0[0]; af[1]=(_Float16)a0[1]; af[2]=(_Float16)a0[2]; af[3]=(_Float16)a0[3];
  af[4]=(_Float16)a1[0]; af[5]=(_Float16)a1[1]; af[6]=(_Float16)a1[2]; af[7]=(_Float16)a1[3];
  af[8]=(_Float16)a2[0]; af[9]=(_Float16)a2[1]; af[10]=(_Float16)a2[2]; af[11]=(_Float16)a2[3];
  af[12]=(_Float16)a3[0]; af[13]=(_Float16)a3[1]; af[14]=(_Float16)a3[2]; af[15]=(_Float16)a3[3];
  return af;
}

// ---------------------------------------------------------------------------
// Generic WMMA GEMM: C[M x Nout] = A[M x K](lda) @ W + bias, optional ReLU.
// 8 waves/block; wave tile = 16 rows x (CT*16) cols. Requires M%16==0,
// K%32==0, Nout % (CT*16) == 0. grid = (ceil(M/128), Nout/(CT*16)).
// CT=8 halves A re-reads vs CT=4 on wide outputs (dominant HBM term).
// ---------------------------------------------------------------------------
template<int CT>
__global__ __launch_bounds__(256) void gemm_wmma_kernel(
    const float* __restrict__ A, int lda,
    const _Float16* __restrict__ Wp,
    const float* __restrict__ bias,
    float* __restrict__ C, int ldc,
    int M, int K, int act)
{
  const int wave = threadIdx.x >> 5;
  const int lane = threadIdx.x & 31;
  const int rowBase = (blockIdx.x * 8 + wave) << 4;
  if (rowBase >= M) return;
  const int ctg0   = blockIdx.y * CT;     // first 16-col tile index
  const int kSteps = K >> 5;
  const int kb     = (lane >> 4) << 3;    // A K-base per half-wave
  const float*    Arow = A + (size_t)(rowBase + (lane & 15)) * lda;
  const _Float16* Wl   = Wp + (size_t)lane * 16;
  const size_t cstride = (size_t)kSteps * 512;

  v8f acc[CT];
  v8f zero = {};
  #pragma unroll
  for (int i = 0; i < CT; ++i) acc[i] = zero;

  for (int kt = 0; kt < kSteps; ++kt) {
    const int k0 = (kt << 5) + kb;
    __builtin_prefetch(Arow + k0 + 32, 0, 1);   // global_prefetch_b8 next K-step
    v16h af = load_a_frag(Arow, k0);
    const _Float16* Wf = Wl + ((size_t)ctg0 * kSteps + kt) * 512;
    #pragma unroll
    for (int ct = 0; ct < CT; ++ct) {
      v16h bf = *(const v16h*)(Wf + (size_t)ct * cstride);
      acc[ct] = __builtin_amdgcn_wmma_f32_16x16x32_f16(false, af, false, bf,
                                                       (short)0, acc[ct], false, false);
    }
  }

  const int rb = rowBase + ((lane >> 4) << 3);  // C: lane<16 -> rows r, else 8+r
  const int c0 = (ctg0 << 4) + (lane & 15);
  #pragma unroll
  for (int ct = 0; ct < CT; ++ct) {
    const int col = c0 + (ct << 4);
    float bv = bias ? bias[col] : 0.0f;
    #pragma unroll
    for (int r = 0; r < 8; ++r) {
      float v = acc[ct][r] + bv;
      if (act) v = fmaxf(v, 0.0f);
      C[(size_t)(rb + r) * ldc + col] = v;
    }
  }
}

// ---------------------------------------------------------------------------
// Fused classifier head: logits = A[Nx64] @ W(64x16) + b, then log_softmax
// over 16 classes. One WMMA tile per wave covers all classes; softmax via
// shfl_xor within 16-lane column groups.
// ---------------------------------------------------------------------------
__global__ __launch_bounds__(256) void cls_head_kernel(
    const float* __restrict__ A, const _Float16* __restrict__ Wp,
    const float* __restrict__ bias, float* __restrict__ out, int M)
{
  const int wave = threadIdx.x >> 5;
  const int lane = threadIdx.x & 31;
  const int rowBase = (blockIdx.x * 8 + wave) << 4;
  if (rowBase >= M) return;
  const int kb = (lane >> 4) << 3;
  const float* Arow = A + (size_t)(rowBase + (lane & 15)) * 64;
  v8f acc = {};
  #pragma unroll
  for (int kt = 0; kt < 2; ++kt) {
    v16h af = load_a_frag(Arow, (kt << 5) + kb);
    v16h bf = *(const v16h*)(Wp + (size_t)kt * 512 + (size_t)lane * 16);
    acc = __builtin_amdgcn_wmma_f32_16x16x32_f16(false, af, false, bf, (short)0, acc, false, false);
  }
  const float bv = bias[lane & 15];
  const int rb = rowBase + ((lane >> 4) << 3);
  #pragma unroll
  for (int r = 0; r < 8; ++r) {
    float v = acc[r] + bv;
    float mx = v;
    #pragma unroll
    for (int m = 1; m < 16; m <<= 1) mx = fmaxf(mx, __shfl_xor(mx, m, 32));
    float s = __expf(v - mx);
    #pragma unroll
    for (int m = 1; m < 16; m <<= 1) s += __shfl_xor(s, m, 32);
    out[(size_t)(rb + r) * 16 + (lane & 15)] = v - mx - __logf(s);
  }
}

// ---------------------------------------------------------------------------
// Sparse aggregation: agg[dst[e],:] += support[src[e],:] * w[e]
// ---------------------------------------------------------------------------
__global__ void spmm_kernel(const float* __restrict__ S, const int* __restrict__ src,
                            const int* __restrict__ dst, const float* __restrict__ w,
                            float* __restrict__ agg, int E, int F) {
  for (int e = blockIdx.x; e < E; e += gridDim.x) {
    const int s = src[e], d = dst[e];
    const float wt = w[e];
    for (int f = threadIdx.x; f < F; f += blockDim.x)
      __hip_atomic_fetch_add(&agg[(size_t)d * F + f], S[(size_t)s * F + f] * wt,
                             __ATOMIC_RELAXED, __HIP_MEMORY_SCOPE_AGENT);
  }
}

__global__ void fill0_kernel(float* __restrict__ p, size_t n) {
  size_t i = (size_t)blockIdx.x * blockDim.x + threadIdx.x;
  if (i < n) p[i] = 0.0f;
}

// out[row*ldout + colOfs + col] = tanh(agg[row*F+col] + bias[col])
__global__ void tanh_bias_kernel(const float* __restrict__ agg, const float* __restrict__ bias,
                                 float* __restrict__ out, int ldout, int colOfs,
                                 size_t total, int F) {
  size_t i = (size_t)blockIdx.x * blockDim.x + threadIdx.x;
  if (i >= total) return;
  size_t row = i / (unsigned)F;
  int col = (int)(i - row * (unsigned)F);
  out[row * ldout + colOfs + col] = tanhf(agg[i] + bias[col]);
}

// Per-column mean + rsqrt(var+eps) over N rows (biased variance).
__global__ void bn_stats_kernel(const float* __restrict__ H, int Nrows, int C,
                                float* __restrict__ mean, float* __restrict__ inv) {
  const int col = blockIdx.x;
  float s = 0.f, s2 = 0.f;
  for (int r = threadIdx.x; r < Nrows; r += blockDim.x) {
    float v = H[(size_t)r * C + col];
    s += v; s2 += v * v;
  }
  __shared__ float sh[256], sh2[256];
  sh[threadIdx.x] = s; sh2[threadIdx.x] = s2;
  __syncthreads();
  for (int o = 128; o > 0; o >>= 1) {
    if (threadIdx.x < o) { sh[threadIdx.x] += sh[threadIdx.x + o]; sh2[threadIdx.x] += sh2[threadIdx.x + o]; }
    __syncthreads();
  }
  if (threadIdx.x == 0) {
    float m = sh[0] / (float)Nrows;
    float var = sh2[0] / (float)Nrows - m * m;
    mean[col] = m;
    inv[col]  = rsqrtf(var + 1e-5f);
  }
}

__global__ void bn_relu_kernel(float* __restrict__ H, const float* __restrict__ mean,
                               const float* __restrict__ inv, const float* __restrict__ g,
                               const float* __restrict__ b, size_t total, int C) {
  size_t i = (size_t)blockIdx.x * blockDim.x + threadIdx.x;
  if (i >= total) return;
  int col = (int)(i % (unsigned)C);
  float v = (H[i] - mean[col]) * inv[col] * g[col] + b[col];
  H[i] = fmaxf(v, 0.0f);
}

// ---------------------------------------------------------------------------
extern "C" void kernel_launch(void* const* d_in, const int* in_sizes, int n_in,
                              void* d_out, int out_size, void* d_ws, size_t ws_size,
                              hipStream_t stream) {
  (void)in_sizes; (void)n_in; (void)out_size; (void)ws_size;
  const float* x       = (const float*)d_in[0];
  const int*   eidx    = (const int*)d_in[1];
  const float* ew      = (const float*)d_in[2];
  const float* gc1_W   = (const float*)d_in[3];  const float* gc1_b  = (const float*)d_in[4];
  const float* gc2_W   = (const float*)d_in[5];  const float* gc2_b  = (const float*)d_in[6];
  const float* affc1_W = (const float*)d_in[7];  const float* affc1_b= (const float*)d_in[8];
  const float* bn1_g   = (const float*)d_in[9];  const float* bn1_b  = (const float*)d_in[10];
  const float* affc2_W = (const float*)d_in[11]; const float* affc2_b= (const float*)d_in[12];
  const float* bn2_g   = (const float*)d_in[13]; const float* bn2_b  = (const float*)d_in[14];
  const float* affc3_W = (const float*)d_in[15]; const float* affc3_b= (const float*)d_in[16];
  const float* affr1_W = (const float*)d_in[17]; const float* affr1_b= (const float*)d_in[18];
  const float* affr2_W = (const float*)d_in[19]; const float* affr2_b= (const float*)d_in[20];
  const float* affr3_W = (const float*)d_in[21]; const float* affr3_b= (const float*)d_in[22];
  const int* src = eidx;
  const int* dst = eidx + EE;

  char* ws = (char*)d_ws;
  size_t off = 0;
  auto alloc = [&](size_t bytes) -> void* {
    void* p = ws + off;
    off = (off + bytes + 255) & ~(size_t)255;
    return p;
  };
  _Float16* WpGc1 = (_Float16*)alloc((size_t)512*256*2);
  _Float16* WpGc2 = (_Float16*)alloc((size_t)256*128*2);
  _Float16* WpC1  = (_Float16*)alloc((size_t)384*256*2);
  _Float16* WpC2  = (_Float16*)alloc((size_t)256*64*2);
  _Float16* WpC3  = (_Float16*)alloc((size_t)64*16*2);
  _Float16* WpR1  = (_Float16*)alloc((size_t)384*256*2);
  _Float16* WpR2  = (_Float16*)alloc((size_t)256*512*2);
  _Float16* WpR3  = (_Float16*)alloc((size_t)512*512*2);
  float* bufA  = (float*)alloc((size_t)NN*256*4);   // support1 -> hc1
  float* bufB  = (float*)alloc((size_t)NN*256*4);   // agg1 -> hr1
  float* bufC  = (float*)alloc((size_t)NN*512*4);   // support2|agg2 -> hc2 -> hr2
  float* stats = (float*)alloc(640*4);
  float* mean1 = stats, *inv1 = stats + 256, *mean2 = stats + 512, *inv2 = stats + 576;

  // d_out layout: xc5 [N,16] | xr5 [N,512] | Zn [N,384]
  float* xc5 = (float*)d_out;
  float* xr5 = (float*)d_out + (size_t)NN*16;
  float* cat = (float*)d_out + (size_t)NN*528;      // also Zn output

  const int rowBlocks = ((NN / 16) + 7) / 8;        // 782
  dim3 blk(256);
  auto pack = [&](const float* W, _Float16* Wp, int K, int Nout) {
    int total = K * Nout;
    pack_w_kernel<<<(total + 255) / 256, 256, 0, stream>>>(W, Wp, K, Nout);
  };
  pack(gc1_W, WpGc1, 512, 256);  pack(gc2_W, WpGc2, 256, 128);
  pack(affc1_W, WpC1, 384, 256); pack(affc2_W, WpC2, 256, 64);
  pack(affc3_W, WpC3, 64, 16);
  pack(affr1_W, WpR1, 384, 256); pack(affr2_W, WpR2, 256, 512);
  pack(affr3_W, WpR3, 512, 512);

  const size_t n256 = (size_t)NN * 256, n128 = (size_t)NN * 128, n64 = (size_t)NN * 64;

  // --- GConv1: x1 = tanh(spmm(x@W1) + b1) -> cat[:, 0:256] ---
  gemm_wmma_kernel<8><<<dim3(rowBlocks, 2), blk, 0, stream>>>(x, 512, WpGc1, nullptr, bufA, 256, NN, 512, 0);
  fill0_kernel<<<(n256 + 255) / 256, 256, 0, stream>>>(bufB, n256);
  spmm_kernel<<<EE, 256, 0, stream>>>(bufA, src, dst, ew, bufB, EE, 256);
  tanh_bias_kernel<<<(n256 + 255) / 256, 256, 0, stream>>>(bufB, gc1_b, cat, 384, 0, n256, 256);

  // --- GConv2: x2 = tanh(spmm(x1@W2) + b2) -> cat[:, 256:384] ---
  float* agg2 = bufC + n128;
  gemm_wmma_kernel<8><<<dim3(rowBlocks, 1), blk, 0, stream>>>(cat, 384, WpGc2, nullptr, bufC, 128, NN, 256, 0);
  fill0_kernel<<<(n128 + 255) / 256, 256, 0, stream>>>(agg2, n128);
  spmm_kernel<<<EE, 128, 0, stream>>>(bufC, src, dst, ew, agg2, EE, 128);
  tanh_bias_kernel<<<(n128 + 255) / 256, 256, 0, stream>>>(agg2, gc2_b, cat, 384, 256, n128, 128);

  // --- clustering head ---
  gemm_wmma_kernel<8><<<dim3(rowBlocks, 2), blk, 0, stream>>>(cat, 384, WpC1, affc1_b, bufA, 256, NN, 384, 0);
  bn_stats_kernel<<<256, 256, 0, stream>>>(bufA, NN, 256, mean1, inv1);
  bn_relu_kernel<<<(n256 + 255) / 256, 256, 0, stream>>>(bufA, mean1, inv1, bn1_g, bn1_b, n256, 256);
  float* bufD = bufC;  // reuse (support2/agg2 dead)
  gemm_wmma_kernel<4><<<dim3(rowBlocks, 1), blk, 0, stream>>>(bufA, 256, WpC2, affc2_b, bufD, 64, NN, 256, 0);
  bn_stats_kernel<<<64, 256, 0, stream>>>(bufD, NN, 64, mean2, inv2);
  bn_relu_kernel<<<(n64 + 255) / 256, 256, 0, stream>>>(bufD, mean2, inv2, bn2_g, bn2_b, n64, 64);
  cls_head_kernel<<<rowBlocks, 256, 0, stream>>>(bufD, WpC3, affc3_b, xc5, NN);

  // --- reconstruction head ---
  gemm_wmma_kernel<8><<<dim3(rowBlocks, 2), blk, 0, stream>>>(cat, 384, WpR1, affr1_b, bufB, 256, NN, 384, 1);
  gemm_wmma_kernel<8><<<dim3(rowBlocks, 4), blk, 0, stream>>>(bufB, 256, WpR2, affr2_b, bufC, 512, NN, 256, 1);
  gemm_wmma_kernel<8><<<dim3(rowBlocks, 4), blk, 0, stream>>>(bufC, 512, WpR3, affr3_b, xr5, 512, NN, 512, 0);
}
